// InstructionsModel_6227702579420
// MI455X (gfx1250) — matrix operand order repaired and checked by reading, compiled, not run on hardware
//
#include <hip/hip_runtime.h>

typedef __attribute__((ext_vector_type(16))) __bf16 v16bf;
typedef __attribute__((ext_vector_type(8)))  float  v8f;

// ---------- bf16 helpers ----------
__device__ __forceinline__ unsigned short pk1(float a) {
  unsigned ua = __float_as_uint(a);
  return (unsigned short)((ua + 0x7FFFu + ((ua >> 16) & 1u)) >> 16);
}
__device__ __forceinline__ float bf2f(unsigned short u) {
  return __uint_as_float(((unsigned)u) << 16);
}

// ---------- WMMA fragment loaders (contiguous K=16h+s order, A and B alike) ----------
// All sources are pre-converted bf16, K-contiguous => pure wide loads.
// A 16x32: lane M=lane%16, h=lane/16. K=16h+2r => 8 consecutive u32 at 8h.
__device__ __forceinline__ v16bf ldA(const unsigned short* p, int h) {
  union { unsigned u[8]; v16bf v; } x;
  const unsigned* q = (const unsigned*)p;
#pragma unroll
  for (int r = 0; r < 8; ++r) x.u[r] = q[8 * h + r];
  return x.v;
}
// B 32x16: lane N=lane%16, h=lane/16; K=16h+2r => 8 consecutive u32 at 8h.
__device__ __forceinline__ v16bf ldB(const unsigned short* p, int h) {
  union { unsigned u[8]; v16bf v; } x;
  const unsigned* q = (const unsigned*)p + 8 * h;
#pragma unroll
  for (int r = 0; r < 8; ++r) x.u[r] = q[r];
  return x.v;
}
// B with K-stride s (elements) -- only used in tiny k_attn PV.
__device__ __forceinline__ v16bf ldBs(const unsigned short* p, int s, int h) {
  union { unsigned u[8]; v16bf v; } x;
#pragma unroll
  for (int r = 0; r < 8; ++r) {
    int k = 16 * h + 2 * r;
    x.u[r] = (unsigned)p[k * s] | ((unsigned)p[(k + 1) * s] << 16);
  }
  return x.v;
}

__device__ __forceinline__ v8f wmma_bf16(v16bf a, v16bf b, v8f c) {
  return __builtin_amdgcn_wmma_f32_16x16x32_bf16(false, a, false, b, (short)0, c,
                                                 false, false);
}
__device__ __forceinline__ v8f vzero() {
  v8f z = {0.f, 0.f, 0.f, 0.f, 0.f, 0.f, 0.f, 0.f};
  return z;
}
__device__ __forceinline__ float sigm(float x) { return 1.0f / (1.0f + __expf(-x)); }
__device__ __forceinline__ float tanh_(float x) {
  return 1.0f - 2.0f / (__expf(2.0f * x) + 1.0f);
}

// ---------- CDNA5 async copy to LDS (ASYNCcnt-tracked) ----------
__device__ __forceinline__ void async_b128(unsigned lds_off, const void* g) {
  asm volatile("global_load_async_to_lds_b128 %0, %1, off"
               :: "v"(lds_off), "v"(g) : "memory");
}
__device__ __forceinline__ void wait_async0() {
  asm volatile("s_wait_asynccnt 0x0" ::: "memory");
}

#define NEG_INF (-1e30f)

// ======================================================================
// One-shot fp32 -> bf16 conversion kernels
// ======================================================================
__global__ void k_cvt(const float* __restrict__ s, unsigned short* __restrict__ d,
                      int n) {
  int i = blockIdx.x * blockDim.x + threadIdx.x;
  int st = gridDim.x * blockDim.x;
  for (; i < n; i += st) d[i] = pk1(s[i]);
}
// zero-padded variant (pads vocab to 10016 rows so async loads need no guard)
__global__ void k_cvt_pad(const float* __restrict__ s, unsigned short* __restrict__ d,
                          int n, int ntot) {
  int i = blockIdx.x * blockDim.x + threadIdx.x;
  int st = gridDim.x * blockDim.x;
  for (; i < ntot; i += st) d[i] = (i < n) ? pk1(s[i]) : (unsigned short)0;
}
// tagger_weight [e,f] -> transposed bf16 [f,e] (N-major for stride-1 B frags)
__global__ void k_cvt_t(const float* __restrict__ s, unsigned short* __restrict__ d) {
  int i = blockIdx.x * 256 + threadIdx.x;  // 65536
  int e = i >> 8, f = i & 255;
  d[f * 256 + e] = pk1(s[e * 256 + f]);
}

// ======================================================================
// Kernel 1: q_proj[16384,256] = questions @ tagger_weight  -> bf16
// ======================================================================
__global__ __launch_bounds__(256) void k_qproj(const unsigned short* __restrict__ qb,
                                               const unsigned short* __restrict__ twt,
                                               unsigned short* __restrict__ qp) {
  const int tid = threadIdx.x, lane = tid & 31, w = tid >> 5;
  const int nl = lane & 15, h = lane >> 4;
  const int tile = blockIdx.x * 8 + w, Mt = tile >> 4, Nt = tile & 15;
  v8f a = vzero();
#pragma unroll
  for (int f = 0; f < 8; ++f)
    a = wmma_bf16(ldA(qb + (Mt * 16 + nl) * 256 + 32 * f, h),
                  ldB(twt + (Nt * 16 + nl) * 256 + 32 * f, h), a);
#pragma unroll
  for (int i = 0; i < 8; ++i)
    qp[(Mt * 16 + i + 8 * h) * 256 + Nt * 16 + nl] = pk1(a[i]);
}

// ======================================================================
// Kernel 2: streaming-softmax tagger over 10001 keys (flash-style).
// Vocab chunks double-buffered in LDS via GLOBAL_LOAD_ASYNC_TO_LDS_B128:
// chunk c+1 streams in while chunk c is transposed + consumed by WMMA.
// ======================================================================
__global__ __launch_bounds__(256) void k_tagger(
    const unsigned short* __restrict__ qproj, const unsigned short* __restrict__ vocb,
    const float* __restrict__ questions, const float* __restrict__ defemb,
    unsigned short* __restrict__ tagged) {
  __shared__ __align__(16) unsigned short ch[2][32 * 256];  // chunk [key][e] x2
  __shared__ __align__(16) unsigned short chT[256 * 32];    // chunk [e][key]
  __shared__ __align__(16) unsigned short pb[8][16 * 32];   // per-wave P staging
  const int tid = threadIdx.x, lane = tid & 31, w = tid >> 5;
  const int nl = lane & 15, h = lane >> 4;
  const int qt = blockIdx.x * 8 + w;
  const int qrow0 = qt * 16;

  const unsigned ch_off0 = (unsigned)(unsigned long long)(void*)&ch[0][0];
  const unsigned ch_off1 = (unsigned)(unsigned long long)(void*)&ch[1][0];

  // resident A fragments for the 16 query rows (K = 256)
  v16bf qa[8];
#pragma unroll
  for (int f = 0; f < 8; ++f)
    qa[f] = ldA(qproj + (qrow0 + nl) * 256 + 32 * f, h);

  // default-embedding logit per query row (lane nl owns row nl, half-split K)
  float z = 0.f;
  {
    const unsigned short* qr = qproj + (qrow0 + nl) * 256 + h * 128;
#pragma unroll 16
    for (int k = 0; k < 128; ++k) z += bf2f(qr[k]) * defemb[h * 128 + k];
    z += __shfl_xor(z, 16, 32);
  }

  float rm[8], rl[8];
  v8f acc[16];
#pragma unroll
  for (int r = 0; r < 8; ++r) { rm[r] = NEG_INF; rl[r] = 0.f; }
#pragma unroll
  for (int nt = 0; nt < 16; ++nt) acc[nt] = vzero();

  // async-issue one 16KB chunk (1024 x b128, 4 per thread)
  auto issue = [&](int c, unsigned dst_off) {
    const char* src = (const char*)(vocb + c * 32 * 256);
#pragma unroll
    for (int i = 0; i < 4; ++i) {
      int idx = i * 256 + tid;
      async_b128(dst_off + idx * 16, src + idx * 16);
    }
  };

  issue(0, ch_off0);
  for (int c = 0; c < 313; ++c) {              // 313*32 >= 10000 (padded rows 0)
    const int base = c * 32, buf = c & 1;
    wait_async0();                             // my slice of chunk c landed
    __syncthreads();                           // everyone's slice landed;
                                               // also fences last chunk's chT reads
    if (c + 1 < 313) issue(c + 1, buf ? ch_off0 : ch_off1);

    // build transposed copy for the PV operand
    {
      const unsigned* c32 = (const unsigned*)ch[buf];
#pragma unroll 4
      for (int i = 0; i < 16; ++i) {
        int idx = i * 256 + tid;
        unsigned val = c32[idx];
        int row = idx >> 7, e = (idx & 127) * 2;
        chT[e * 32 + row] = (unsigned short)(val & 0xFFFFu);
        chT[(e + 1) * 32 + row] = (unsigned short)(val >> 16);
      }
    }
    __syncthreads();

    // QK^T : two 16x16 score tiles (keys along N) -- B is N-major, stride-1
    v8f s0 = vzero(), s1 = vzero();
#pragma unroll
    for (int f = 0; f < 8; ++f) {
      s0 = wmma_bf16(qa[f], ldB(ch[buf] + (0 + nl) * 256 + 32 * f, h), s0);
      s1 = wmma_bf16(qa[f], ldB(ch[buf] + (16 + nl) * 256 + 32 * f, h), s1);
    }
    if (base + nl >= 10000) {
#pragma unroll
      for (int r = 0; r < 8; ++r) s0[r] = NEG_INF;
    }
    if (base + 16 + nl >= 10000) {
#pragma unroll
      for (int r = 0; r < 8; ++r) s1[r] = NEG_INF;
    }

    // online softmax update
    float scl[8];
#pragma unroll
    for (int r = 0; r < 8; ++r) {
      float mx = fmaxf(s0[r], s1[r]);
#pragma unroll
      for (int d = 8; d >= 1; d >>= 1) mx = fmaxf(mx, __shfl_xor(mx, d, 32));
      float mn = fmaxf(rm[r], mx);
      float p0 = __expf(s0[r] - mn), p1 = __expf(s1[r] - mn);
      float sm = p0 + p1;
#pragma unroll
      for (int d = 8; d >= 1; d >>= 1) sm += __shfl_xor(sm, d, 32);
      float sc = __expf(rm[r] - mn);
      rl[r] = rl[r] * sc + sm;
      rm[r] = mn;
      scl[r] = sc;
      s0[r] = p0; s1[r] = p1;                  // reuse as P
    }
#pragma unroll
    for (int nt = 0; nt < 16; ++nt)
#pragma unroll
      for (int r = 0; r < 8; ++r) acc[nt][r] *= scl[r];

    // stage P (C-layout -> A-layout) through per-wave LDS
#pragma unroll
    for (int r = 0; r < 8; ++r) {
      int row = r + 8 * h;
      pb[w][row * 32 + nl] = pk1(s0[r]);
      pb[w][row * 32 + 16 + nl] = pk1(s1[r]);
    }
    asm volatile("s_wait_dscnt 0x0" ::: "memory");
    v16bf pa = ldA(&pb[w][nl * 32], h);

    // PV: acc[16x256] += P[16x32] @ chunk[32x256] -- B from transposed copy
#pragma unroll
    for (int nt = 0; nt < 16; ++nt)
      acc[nt] = wmma_bf16(pa, ldB(chT + (nt * 16 + nl) * 32, h), acc[nt]);
  }

  // fold in default row (value = the query itself), normalize, store bf16
  float pdv[8], inv[8], scl2[8];
#pragma unroll
  for (int r = 0; r < 8; ++r) {
    float zc = __shfl(z, r + 8 * h, 32);
    float mn = fmaxf(rm[r], zc);
    float s = __expf(rm[r] - mn);
    float pd = __expf(zc - mn);
    float l = rl[r] * s + pd;
    scl2[r] = s; pdv[r] = pd; inv[r] = 1.0f / l;
  }
#pragma unroll
  for (int nt = 0; nt < 16; ++nt)
#pragma unroll
    for (int r = 0; r < 8; ++r) {
      int row = qrow0 + r + 8 * h, col = nt * 16 + nl;
      float qv = questions[row * 256 + col];
      tagged[row * 256 + col] = pk1((acc[nt][r] * scl2[r] + pdv[r] * qv) * inv[r]);
    }
}

// ======================================================================
// Kernel 3: LSTM, 256 steps, 1 WG x 16 waves. c in regs, h in bf16 LDS
// (double buffered), packed-sequence freeze per element.
// ======================================================================
__global__ __launch_bounds__(512) void k_lstm(
    const unsigned short* __restrict__ tagged, const unsigned short* __restrict__ w_ih,
    const unsigned short* __restrict__ w_hh, const float* __restrict__ b_ih,
    const float* __restrict__ b_hh, const int* __restrict__ lengths,
    unsigned short* __restrict__ enc_b, float* __restrict__ enc_out) {
  extern __shared__ __align__(16) unsigned short hsm[];  // 2 x 64*256 bf16
  unsigned short* hb0 = hsm;
  unsigned short* hb1 = hsm + 16384;
  const int tid = threadIdx.x, lane = tid & 31, w = tid >> 5;
  const int nl = lane & 15, h = lane >> 4;

  for (int i = tid; i < 16384; i += 512) hb0[i] = 0;
  __syncthreads();

  v8f cst[4];
#pragma unroll
  for (int q = 0; q < 4; ++q) cst[q] = vzero();

  int cur = 0;
  for (int t = 0; t < 256; ++t) {
    const unsigned short* hc = cur ? hb1 : hb0;
    unsigned short* hn = cur ? hb0 : hb1;
#pragma unroll
    for (int q = 0; q < 4; ++q) {
      int id = w * 4 + q, Mt = id >> 4, fg = id & 15;
      if (t + 1 < 256)  // prefetch next step's x operand (global_prefetch_b8)
        __builtin_prefetch(tagged + (Mt * 16 + nl) * 65536 + (t + 1) * 256, 0, 3);
      v8f g4[4];
#pragma unroll
      for (int gi = 0; gi < 4; ++gi) {
        int n = gi * 256 + fg * 16 + nl;       // gate column in [0,1024)
        float bias = b_ih[n] + b_hh[n];
        v8f a;
#pragma unroll
        for (int i = 0; i < 8; ++i) a[i] = bias;
        const unsigned short* xb = tagged + (Mt * 16 + nl) * 65536 + t * 256;
        const unsigned short* wb = w_ih + n * 256;   // W.T column n, N-major
        const unsigned short* vb = w_hh + n * 256;
#pragma unroll
        for (int f = 0; f < 8; ++f) {
          a = wmma_bf16(ldA(xb + 32 * f, h), ldB(wb + 32 * f, h), a);
          a = wmma_bf16(ldA(hc + (Mt * 16 + nl) * 256 + 32 * f, h),
                        ldB(vb + 32 * f, h), a);
        }
        g4[gi] = a;
      }
#pragma unroll
      for (int i = 0; i < 8; ++i) {
        int b = Mt * 16 + i + 8 * h;
        int col = fg * 16 + nl;
        float ci = cst[q][i];
        float iv = sigm(g4[0][i]), fv = sigm(g4[1][i]);
        float gv = tanh_(g4[2][i]), ov = sigm(g4[3][i]);
        float cn = fv * ci + iv * gv;
        float hv = ov * tanh_(cn);
        bool m = t < lengths[b];
        cst[q][i] = m ? cn : ci;
        hn[b * 256 + col] = m ? pk1(hv) : hc[b * 256 + col];
      }
    }
    __syncthreads();
    cur ^= 1;
  }
  const unsigned short* hf = cur ? hb1 : hb0;
  for (int i = tid; i < 16384; i += 512) {
    enc_b[i] = hf[i];
    enc_out[i] = bf2f(hf[i]);
  }
}

// ======================================================================
// Kernel 4: xpart = encoded @ rnn_w_ih.T + b_ih + b_hh  (fp32 out)
// ======================================================================
__global__ __launch_bounds__(256) void k_dec_prep(
    const unsigned short* __restrict__ enc, const unsigned short* __restrict__ w_ih,
    const float* __restrict__ b_ih, const float* __restrict__ b_hh,
    float* __restrict__ xpart) {
  const int tid = threadIdx.x, lane = tid & 31, w = tid >> 5;
  const int nl = lane & 15, h = lane >> 4;
#pragma unroll
  for (int j = 0; j < 8; ++j) {
    int id = w * 8 + j, Mt = id >> 4, Nt = id & 15, n = Nt * 16 + nl;
    float bias = b_ih[n] + b_hh[n];
    v8f a;
#pragma unroll
    for (int i = 0; i < 8; ++i) a[i] = bias;
#pragma unroll
    for (int f = 0; f < 8; ++f)
      a = wmma_bf16(ldA(enc + (Mt * 16 + nl) * 256 + 32 * f, h),
                    ldB(w_ih + n * 256 + 32 * f, h), a);
#pragma unroll
    for (int i = 0; i < 8; ++i)
      xpart[(Mt * 16 + i + 8 * h) * 256 + n] = a[i];
  }
}

// ======================================================================
// Kernel 5: RNNCell decoder, 8 steps; hiddens stored bf16 [B, I, E]
// ======================================================================
__global__ __launch_bounds__(256) void k_dec(const float* __restrict__ xpart,
                                             const unsigned short* __restrict__ w_hh,
                                             unsigned short* __restrict__ hiddens) {
  extern __shared__ __align__(16) unsigned short hs[];   // 2 x 64*256 bf16
  unsigned short* hb0 = hs;
  unsigned short* hb1 = hs + 16384;
  const int tid = threadIdx.x, lane = tid & 31, w = tid >> 5;
  const int nl = lane & 15, h = lane >> 4;

  for (int i = tid; i < 16384; i += 256) {     // step 0: hx = relu(xpart)
    unsigned short v = pk1(fmaxf(xpart[i], 0.f));
    hb0[i] = v;
    hiddens[((i >> 8) * 8 + 0) * 256 + (i & 255)] = v;
  }
  __syncthreads();
  int cur = 0;
  for (int s = 1; s < 8; ++s) {
    const unsigned short* hc = cur ? hb1 : hb0;
    unsigned short* hn = cur ? hb0 : hb1;
#pragma unroll
    for (int j = 0; j < 8; ++j) {
      int id = w * 8 + j, Mt = id >> 4, Nt = id & 15, n = Nt * 16 + nl;
      v8f a;
#pragma unroll
      for (int i = 0; i < 8; ++i) a[i] = xpart[(Mt * 16 + i + 8 * h) * 256 + n];
#pragma unroll
      for (int f = 0; f < 8; ++f)
        a = wmma_bf16(ldA(hc + (Mt * 16 + nl) * 256 + 32 * f, h),
                      ldB(w_hh + n * 256 + 32 * f, h), a);
#pragma unroll
      for (int i = 0; i < 8; ++i) {
        unsigned short v = pk1(fmaxf(a[i], 0.f));
        int b = Mt * 16 + i + 8 * h;
        hn[b * 256 + n] = v;
        hiddens[(b * 8 + s) * 256 + n] = v;
      }
    }
    __syncthreads();
    cur ^= 1;
  }
}

// ======================================================================
// Kernel 6: masked attention over tagged tokens; one wave per batch row,
// flash-style over S=256 in 8 chunks of 32 keys. Q rows padded 8->16.
// ======================================================================
__global__ __launch_bounds__(32) void k_attn(const unsigned short* __restrict__ hiddens,
                                             const unsigned short* __restrict__ tagged,
                                             const int* __restrict__ lengths,
                                             float* __restrict__ out) {
  __shared__ __align__(16) unsigned short pb[16 * 32];
  const int lane = threadIdx.x & 31, nl = lane & 15, h = lane >> 4;
  const int b = blockIdx.x;
  const int len = lengths[b];
  const unsigned short* tg = tagged + b * 65536;

  v16bf qa[8];
#pragma unroll
  for (int f = 0; f < 8; ++f)
    qa[f] = ldA(hiddens + (b * 8 + (nl & 7)) * 256 + 32 * f, h);

  float rm[8], rl[8];
  v8f acc[16];
#pragma unroll
  for (int r = 0; r < 8; ++r) { rm[r] = NEG_INF; rl[r] = 0.f; }
#pragma unroll
  for (int nt = 0; nt < 16; ++nt) acc[nt] = vzero();

  for (int c = 0; c < 8; ++c) {
    const int base = c * 32;
    v8f s0 = vzero(), s1 = vzero();
#pragma unroll
    for (int f = 0; f < 8; ++f) {
      s0 = wmma_bf16(qa[f], ldB(tg + (base + nl) * 256 + 32 * f, h), s0);
      s1 = wmma_bf16(qa[f], ldB(tg + (base + 16 + nl) * 256 + 32 * f, h), s1);
    }
    if (base + nl >= len) {
#pragma unroll
      for (int r = 0; r < 8; ++r) s0[r] = NEG_INF;
    }
    if (base + 16 + nl >= len) {
#pragma unroll
      for (int r = 0; r < 8; ++r) s1[r] = NEG_INF;
    }
    float scl[8];
#pragma unroll
    for (int r = 0; r < 8; ++r) {
      float mx = fmaxf(s0[r], s1[r]);
#pragma unroll
      for (int d = 8; d >= 1; d >>= 1) mx = fmaxf(mx, __shfl_xor(mx, d, 32));
      float mn = fmaxf(rm[r], mx);
      float p0 = __expf(s0[r] - mn), p1 = __expf(s1[r] - mn);
      float sm = p0 + p1;
#pragma unroll
      for (int d = 8; d >= 1; d >>= 1) sm += __shfl_xor(sm, d, 32);
      float sc = __expf(rm[r] - mn);
      rl[r] = rl[r] * sc + sm;
      rm[r] = mn;
      scl[r] = sc;
      s0[r] = p0; s1[r] = p1;
    }
#pragma unroll
    for (int nt = 0; nt < 16; ++nt)
#pragma unroll
      for (int r = 0; r < 8; ++r) acc[nt][r] *= scl[r];
#pragma unroll
    for (int r = 0; r < 8; ++r) {
      int row = r + 8 * h;
      pb[row * 32 + nl] = pk1(s0[r]);
      pb[row * 32 + 16 + nl] = pk1(s1[r]);
    }
    asm volatile("s_wait_dscnt 0x0" ::: "memory");
    v16bf pa = ldA(&pb[nl * 32], h);
#pragma unroll
    for (int nt = 0; nt < 16; ++nt)
      acc[nt] = wmma_bf16(pa, ldBs(tg + base * 256 + nt * 16 + nl, 256, h), acc[nt]);
  }
  if (h == 0) {                                // rows 8..15 are duplicates
#pragma unroll
    for (int r = 0; r < 8; ++r) {
      float inv = 1.0f / rl[r];
#pragma unroll
      for (int nt = 0; nt < 16; ++nt)
        out[(b * 8 + r) * 256 + nt * 16 + nl] = acc[nt][r] * inv;
    }
  }
}

// ======================================================================
extern "C" void kernel_launch(void* const* d_in, const int* in_sizes, int n_in,
                              void* d_out, int out_size, void* d_ws, size_t ws_size,
                              hipStream_t stream) {
  (void)in_sizes; (void)n_in; (void)out_size; (void)ws_size;
  const float* vocab = (const float*)d_in[0];
  const float* questions = (const float*)d_in[1];
  const int* lengths = (const int*)d_in[2];
  const float* tw = (const float*)d_in[3];
  const float* de = (const float*)d_in[4];
  const float* lw_ih = (const float*)d_in[5];
  const float* lw_hh = (const float*)d_in[6];
  const float* lb_ih = (const float*)d_in[7];
  const float* lb_hh = (const float*)d_in[8];
  const float* rw_ih = (const float*)d_in[9];
  const float* rw_hh = (const float*)d_in[10];
  const float* rb_ih = (const float*)d_in[11];
  const float* rb_hh = (const float*)d_in[12];

  unsigned short* ub = (unsigned short*)d_ws;
  unsigned short* q_b    = ub;                  // 4,194,304
  unsigned short* voc_b  = ub + 4194304;        // 2,564,096 (10016 rows, padded)
  unsigned short* qp_b   = ub + 6758400;        // 4,194,304
  unsigned short* tag_b  = ub + 10952704;       // 4,194,304
  unsigned short* twt_b  = ub + 15147008;       //    65,536
  unsigned short* lwih_b = ub + 15212544;       //   262,144
  unsigned short* lwhh_b = ub + 15474688;       //   262,144
  unsigned short* rwih_b = ub + 15736832;       //    65,536
  unsigned short* rwhh_b = ub + 15802368;       //    65,536
  unsigned short* enc_b  = ub + 15867904;       //    16,384
  unsigned short* hid_b  = ub + 15884288;       //   131,072
  float* xpart = (float*)(ub + 16015360);       //    16,384 fp32

  float* out = (float*)d_out;                   // instructions [64,8,256]
  float* enc_out = out + 64 * 8 * 256;          // encoded [64,256]

  // one-shot bf16 conversions (all WMMA operands become stride-1 bf16)
  k_cvt<<<1024, 256, 0, stream>>>(questions, q_b, 4194304);
  k_cvt_pad<<<1024, 256, 0, stream>>>(vocab, voc_b, 2560000, 2564096);
  k_cvt<<<256, 256, 0, stream>>>(lw_ih, lwih_b, 262144);
  k_cvt<<<256, 256, 0, stream>>>(lw_hh, lwhh_b, 262144);
  k_cvt<<<64, 256, 0, stream>>>(rw_ih, rwih_b, 65536);
  k_cvt<<<64, 256, 0, stream>>>(rw_hh, rwhh_b, 65536);
  k_cvt_t<<<256, 256, 0, stream>>>(tw, twt_b);

  k_qproj<<<2048, 256, 0, stream>>>(q_b, twt_b, qp_b);
  k_tagger<<<128, 256, 0, stream>>>(qp_b, voc_b, questions, de, tag_b);
  k_lstm<<<1, 512, 2 * 16384 * sizeof(unsigned short), stream>>>(
      tag_b, lwih_b, lwhh_b, lb_ih, lb_hh, lengths, enc_b, enc_out);
  k_dec_prep<<<1, 256, 0, stream>>>(enc_b, rwih_b, rb_ih, rb_hh, xpart);
  k_dec<<<1, 256, 2 * 16384 * sizeof(unsigned short), stream>>>(xpart, rwhh_b, hid_b);
  k_attn<<<64, 32, 0, stream>>>(hid_b, tag_b, lengths, out);
}
